// MultiLayerSNN_40750649705152
// MI455X (gfx1250) — matrix operand order repaired and checked by reading, not compiled, not run on hardware
//
#include <hip/hip_runtime.h>

typedef __attribute__((ext_vector_type(16))) _Float16 v16h;
typedef __attribute__((ext_vector_type(8)))  _Float16 v8h;
typedef __attribute__((ext_vector_type(8)))  float    v8f;

union AFrag { v16h v; v8h h[2]; };

// ---------------- LDS layout (one 256-thread workgroup per WGP) ----------------
constexpr unsigned OFF_W2B  = 0;                                  // 131072 B
constexpr unsigned OFF_W3B  = OFF_W2B + 16u * 256u * 16u * 2u;    // 131072
constexpr unsigned OFF_CUR1 = OFF_W3B + 8u * 256u * 16u * 2u;     // cur1T f16 [n(256)][m(16)] per pair
constexpr unsigned OFF_SPK1 = OFF_CUR1 + 4u * 16u * 256u * 2u;    // 229376
constexpr unsigned SPK_ROW  = 264;                                 // 256 + 8 pad halves
constexpr unsigned SPKBUF   = 16u * SPK_ROW * 2u;                  // 8448 B
constexpr unsigned OFF_SPK2 = OFF_SPK1 + 4u * SPKBUF;             // 263168
constexpr unsigned OFF_RED  = OFF_SPK2 + 4u * SPKBUF;             // 296960
constexpr unsigned LDS_BYTES = OFF_RED + 4u * 48u * 4u;           // 297728 B < 320K

__global__ void __launch_bounds__(256, 1)
snn_pair_kernel(const float* __restrict__ x,
                const float* __restrict__ W1, const float* __restrict__ b1,
                const float* __restrict__ W2, const float* __restrict__ b2,
                const float* __restrict__ W3, const float* __restrict__ b3,
                const float* __restrict__ Wo, const float* __restrict__ bo,
                const int* __restrict__ nsteps,
                float* __restrict__ out)
{
    extern __shared__ char smem[];
    _Float16* w2B = (_Float16*)(smem + OFF_W2B);
    _Float16* w3B = (_Float16*)(smem + OFF_W3B);

    const int tid  = threadIdx.x;
    const int wave = tid >> 5;
    const int lane = tid & 31;
    const int pair = wave >> 1;   // 4 pairs; each pair owns 16 batch rows
    const int half = wave & 1;    // n-tile split between the waves of a pair

    _Float16* cur1T = (_Float16*)(smem + OFF_CUR1 + (unsigned)pair * 8192u);
    _Float16* spk1  = (_Float16*)(smem + OFF_SPK1 + (unsigned)pair * SPKBUF);
    _Float16* spk2  = (_Float16*)(smem + OFF_SPK2 + (unsigned)pair * SPKBUF);
    unsigned short* spk1U = (unsigned short*)spk1;
    unsigned short* spk2U = (unsigned short*)spk2;
    float*    redS  = (float*)(smem + OFF_RED + (unsigned)pair * 192u);

    // ---- stage W2/W3 into LDS as f16 B-operand tiles: element (k,n) -> [n>>4][k>>5][lane=(n&15)+16*((k>>3)&1)][(k&7)+8*((k>>4)&1)] ----
    for (int i = tid; i < 256 * 256; i += 256) {
        int n = i >> 8, k = i & 255;
        w2B[(((n >> 4) * 256 + (k & ~31) + ((k >> 3) & 1) * 16 + (n & 15)) << 4) + (k & 7) + ((k >> 4) & 1) * 8] = (_Float16)W2[i];
    }
    for (int i = tid; i < 128 * 256; i += 256) {
        int n = i >> 8, k = i & 255;
        w3B[(((n >> 4) * 256 + (k & ~31) + ((k >> 3) & 1) * 16 + (n & 15)) << 4) + (k & 7) + ((k >> 4) & 1) * 8] = (_Float16)W3[i];
    }

    const int rowBase = blockIdx.x * 64 + pair * 16;
    const int col  = lane & 15;   // C-layout: N within tile
    const int hi   = lane >> 4;   // C-layout: 0 -> rows 0-7, 1 -> rows 8-15
    const int arow = lane & 15;   // A-layout: M
    const int ak8  = hi * 8;      // A-layout: K sub-offset

    // ---- cur1 = x @ W1^T + b1, stored f16 transposed [n][m] for vector re-reads ----
    {
        float xr[8][6];
        #pragma unroll
        for (int r = 0; r < 8; ++r) {
            const float* xp = x + (size_t)(rowBase + r + 8 * hi) * 6;
            #pragma unroll
            for (int i = 0; i < 6; ++i) xr[r][i] = xp[i];
        }
        #pragma unroll
        for (int ntl = 0; ntl < 8; ++ntl) {
            int n = half * 128 + ntl * 16 + col;
            float w[6];
            #pragma unroll
            for (int i = 0; i < 6; ++i) w[i] = W1[n * 6 + i];
            float bb = b1[n];
            #pragma unroll
            for (int r = 0; r < 8; ++r) {
                float s = bb;
                #pragma unroll
                for (int i = 0; i < 6; ++i) s += xr[r][i] * w[i];
                cur1T[(unsigned)n * 16u + (r + 8 * hi)] = (_Float16)s;
            }
        }
    }

    // ---- per-lane constant tables (registers) ----
    float bias2r[8], bias3r[4], woR[4][3];
    #pragma unroll
    for (int ntl = 0; ntl < 8; ++ntl) bias2r[ntl] = b2[half * 128 + ntl * 16 + col];
    #pragma unroll
    for (int ntl = 0; ntl < 4; ++ntl) {
        int n = half * 64 + ntl * 16 + col;
        bias3r[ntl] = b3[n];
        #pragma unroll
        for (int j = 0; j < 3; ++j) woR[ntl][j] = Wo[j * 128 + n];
    }

    // ---- persistent membrane state in VGPRs ----
    v8f mem1[8], mem2[8], mem3[4];
    #pragma unroll
    for (int ntl = 0; ntl < 8; ++ntl)
        #pragma unroll
        for (int r = 0; r < 8; ++r) { mem1[ntl][r] = 0.0f; mem2[ntl][r] = 0.0f; }
    #pragma unroll
    for (int ntl = 0; ntl < 4; ++ntl)
        #pragma unroll
        for (int r = 0; r < 8; ++r) mem3[ntl][r] = 0.0f;

    float acc[8][3];
    #pragma unroll
    for (int r = 0; r < 8; ++r)
        #pragma unroll
        for (int j = 0; j < 3; ++j) acc[r][j] = 0.0f;

    const float THRv = 1.0f;
    const unsigned short ONE_H = 0x3C00;  // f16 1.0 bit pattern
    const int T = *nsteps;

    // flat B-fragment bases: fragment f = ntl*8+kt lives at base + f*512 halves
    const _Float16* b2base = w2B + ((unsigned)half * 8u * 256u + (unsigned)lane) * 16u;
    const _Float16* b3base = w3B + ((unsigned)half * 4u * 256u + (unsigned)lane) * 16u;

    __syncthreads();  // weights + cur1 staged

    for (int t = 0; t < T; ++t) {
        // ---- Layer 1 LIF -> spk1 (f16 bit pattern selected, no cvt) ----
        #pragma unroll
        for (int ntl = 0; ntl < 8; ++ntl) {
            int n = half * 128 + ntl * 16 + col;
            v8h cv = *(const v8h*)(cur1T + (unsigned)n * 16u + 8 * hi);
            #pragma unroll
            for (int r = 0; r < 8; ++r) {
                float m = mem1[ntl][r];
                float rst = (m > THRv) ? THRv : 0.0f;
                m = 0.9f * m + (float)cv[r] - rst;
                mem1[ntl][r] = m;
                spk1U[(unsigned)(r + 8 * hi) * SPK_ROW + n] = (m > THRv) ? ONE_H : 0;
            }
        }
        __syncthreads();  // both halves of spk1 visible

        // ---- Layer 2: spk1 @ W2^T on this wave's 8 n-tiles (zero-init C, B pipelined) ----
        AFrag a1[8];
        #pragma unroll
        for (int kt = 0; kt < 8; ++kt) {
            a1[kt].h[0] = *(const v8h*)(spk1 + (unsigned)arow * SPK_ROW + kt * 32 + ak8);
            a1[kt].h[1] = *(const v8h*)(spk1 + (unsigned)arow * SPK_ROW + kt * 32 + 16 + ak8);
        }
        {
            v16h bcur = *(const v16h*)(b2base);
            #pragma unroll
            for (int ntl = 0; ntl < 8; ++ntl) {
                v8f c = {};  // inline-0 C operand; bias folded into LIF below
                #pragma unroll
                for (int kt = 0; kt < 8; ++kt) {
                    const int f = ntl * 8 + kt + 1;
                    v16h bnext;
                    if (f < 64) bnext = *(const v16h*)(b2base + (unsigned)f * 512u);
                    c = __builtin_amdgcn_wmma_f32_16x16x32_f16(false, a1[kt].v, false, bcur,
                                                               (short)0, c, false, false);
                    if (f < 64) bcur = bnext;
                }
                #pragma unroll
                for (int r = 0; r < 8; ++r) {
                    float m = mem2[ntl][r];
                    float rst = (m > THRv) ? THRv : 0.0f;
                    m = 0.85f * m + (c[r] + bias2r[ntl]) - rst;
                    mem2[ntl][r] = m;
                    spk2U[(unsigned)(r + 8 * hi) * SPK_ROW + half * 128 + ntl * 16 + col] =
                        (m > THRv) ? ONE_H : 0;
                }
            }
        }
        __syncthreads();  // both halves of spk2 visible

        // ---- Layer 3 + output head (spk3 stays in registers) ----
        AFrag a2[8];
        #pragma unroll
        for (int kt = 0; kt < 8; ++kt) {
            a2[kt].h[0] = *(const v8h*)(spk2 + (unsigned)arow * SPK_ROW + kt * 32 + ak8);
            a2[kt].h[1] = *(const v8h*)(spk2 + (unsigned)arow * SPK_ROW + kt * 32 + 16 + ak8);
        }
        {
            v16h bcur = *(const v16h*)(b3base);
            #pragma unroll
            for (int ntl = 0; ntl < 4; ++ntl) {
                v8f c = {};  // inline-0 C operand; bias folded into LIF below
                #pragma unroll
                for (int kt = 0; kt < 8; ++kt) {
                    const int f = ntl * 8 + kt + 1;
                    v16h bnext;
                    if (f < 32) bnext = *(const v16h*)(b3base + (unsigned)f * 512u);
                    c = __builtin_amdgcn_wmma_f32_16x16x32_f16(false, a2[kt].v, false, bcur,
                                                               (short)0, c, false, false);
                    if (f < 32) bcur = bnext;
                }
                #pragma unroll
                for (int r = 0; r < 8; ++r) {
                    float m = mem3[ntl][r];
                    float rst = (m > THRv) ? THRv : 0.0f;
                    m = 0.8f * m + (c[r] + bias3r[ntl]) - rst;
                    mem3[ntl][r] = m;
                    float s = (m > THRv) ? 1.0f : 0.0f;
                    #pragma unroll
                    for (int j = 0; j < 3; ++j) acc[r][j] += s * woR[ntl][j];
                }
            }
        }
    }

    // ---- reduce output head: over 16 col-lanes, then across the wave pair ----
    #pragma unroll
    for (int r = 0; r < 8; ++r)
        #pragma unroll
        for (int j = 0; j < 3; ++j) {
            float v = acc[r][j];
            v += __shfl_xor(v, 1, 32);
            v += __shfl_xor(v, 2, 32);
            v += __shfl_xor(v, 4, 32);
            v += __shfl_xor(v, 8, 32);
            acc[r][j] = v;  // identical across each 16-lane group
        }
    if (half == 0 && col == 0) {
        #pragma unroll
        for (int r = 0; r < 8; ++r)
            #pragma unroll
            for (int j = 0; j < 3; ++j)
                redS[(hi * 8 + r) * 3 + j] = acc[r][j];
    }
    __syncthreads();
    if (half == 1 && col == 0) {
        const float inv = 1.0f / (float)T;
        #pragma unroll
        for (int r = 0; r < 8; ++r) {
            float* op = out + (size_t)(rowBase + hi * 8 + r) * 3;
            #pragma unroll
            for (int j = 0; j < 3; ++j)
                op[j] = (acc[r][j] + redS[(hi * 8 + r) * 3 + j]) * inv + bo[j];
        }
    }
}

extern "C" void kernel_launch(void* const* d_in, const int* in_sizes, int n_in,
                              void* d_out, int out_size, void* d_ws, size_t ws_size,
                              hipStream_t stream) {
    const float* x  = (const float*)d_in[0];
    const float* W1 = (const float*)d_in[1];
    const float* b1 = (const float*)d_in[2];
    const float* W2 = (const float*)d_in[3];
    const float* b2 = (const float*)d_in[4];
    const float* W3 = (const float*)d_in[5];
    const float* b3 = (const float*)d_in[6];
    const float* Wo = (const float*)d_in[7];
    const float* bo = (const float*)d_in[8];
    const int*   ns = (const int*)d_in[9];
    float* out = (float*)d_out;

    const int batch  = in_sizes[0] / 6;   // x is (B, 6)
    const int blocks = batch / 64;        // 64 rows per block (4 pairs x 16 rows)

    (void)hipFuncSetAttribute((const void*)snn_pair_kernel,
                              hipFuncAttributeMaxDynamicSharedMemorySize,
                              (int)LDS_BYTES);
    snn_pair_kernel<<<blocks, 256, LDS_BYTES, stream>>>(
        x, W1, b1, W2, b2, W3, b3, Wo, bo, ns, out);
}
